// BatchedHGNNLayer_89739046683202
// MI455X (gfx1250) — matrix operand order repaired and checked by reading, not compiled, not run on hardware
//
#include <hip/hip_runtime.h>
#include <hip/hip_bf16.h>

// ---------------------------------------------------------------------------
// BatchedHGNNLayer on MI455X (gfx1250): bf16 WMMA pipeline + TDM-staged
// B-panels in LDS (double buffered, s_wait_tensorcnt synchronized).
// Shapes fixed by the reference: B=8, N=4096, E=2048, C=128 (=C_in=C_out).
// ---------------------------------------------------------------------------

#define BB   8
#define NN   4096
#define EE   2048
#define CC   128
#define KCH  128          // K-chunk staged per TDM transfer (bf16 panel 128x128 = 32KB)
#define EPSV 1e-6f

typedef __attribute__((ext_vector_type(8)))  __bf16       v8bf;
typedef __attribute__((ext_vector_type(16))) __bf16       v16bf;
typedef __attribute__((ext_vector_type(8)))  float        v8f;
typedef __attribute__((ext_vector_type(4)))  unsigned int v4u;
typedef __attribute__((ext_vector_type(8)))  int          v8i;
typedef __attribute__((ext_vector_type(4)))  int          v4i;

// ---- WMMA fragment loaders ------------------------------------------------
// A-matrix (16x32 bf16, row-major [M,K] source):
//   lane L<16 : row=L,    K = {0..7, 16..23}
//   lane L>=16: row=L-16, K = {8..15, 24..31}
// rowp must already point at  row*ld + k0 + koff  (koff = (lane>>4)*8).
__device__ __forceinline__ v16bf load_a_frag(const __bf16* rowp) {
    union { v16bf v; v8bf h[2]; } u;
    u.h[0] = *(const v8bf*)(rowp);
    u.h[1] = *(const v8bf*)(rowp + 16);
    return u.v;
}
// B-matrix (32x16 bf16) loaded from B^T row-major [Ndim,K]:
//   lane L<16 : col=L,    K = {0..7, 16..23}
//   lane L>=16: col=L-16, K = {8..15, 24..31}
// colp must already point at  col*ld + k0 + (lane>>4)*8.
__device__ __forceinline__ v16bf load_b_frag(const __bf16* colp) {
    union { v16bf v; v8bf h[2]; } u;
    u.h[0] = *(const v8bf*)(colp);
    u.h[1] = *(const v8bf*)(colp + 16);
    return u.v;
}

__device__ __forceinline__ v8f wmma_bf16(v16bf a, v16bf b, v8f c) {
    return __builtin_amdgcn_wmma_f32_16x16x32_bf16(
        /*neg_a=*/false, a, /*neg_b=*/false, b,
        /*c_mod=*/(short)0, c, /*reuse_a=*/false, /*reuse_b=*/false);
}

// ---- Tensor Data Mover: 2-D bf16 tile (tile_d1 rows x tile_d0 elems) ------
// Loads from global (row stride = stride_d0 elements) into LDS at lds_off,
// packed row-major [tile_d1][tile_d0]. Issue from ONE wave only.
__device__ __forceinline__ void tdm_load_2d_bf16(unsigned lds_off, const void* gaddr,
                                                 unsigned tensor_d0, unsigned tensor_d1,
                                                 unsigned tile_d0, unsigned tile_d1,
                                                 unsigned stride_d0) {
    unsigned long long ga = (unsigned long long)gaddr;
    v4u g0;
    g0[0] = 1u;                                     // count=1 (valid), user mode
    g0[1] = lds_off;                                // lds_addr (bytes)
    g0[2] = (unsigned)(ga & 0xffffffffu);           // global_addr[31:0]  (bits 95:64)
    g0[3] = (unsigned)((ga >> 32) & 0x01ffffffu)    // global_addr[56:32] (bits 120:96)
          | (2u << 30);                             // type = 2 ("image")
    v8i g1;
    g1[0] = (int)(1u << 16);                        // wg_mask=0, data_size=1 (2 bytes)
    g1[1] = (int)((tensor_d0 & 0xffffu) << 16);     // tensor_dim0[15:0]  @ bits 63:48
    g1[2] = (int)(((tensor_d0 >> 16) & 0xffffu)     // tensor_dim0[31:16] @ bits 79:64
          |       ((tensor_d1 & 0xffffu) << 16));   // tensor_dim1[15:0]  @ bits 95:80
    g1[3] = (int)(((tensor_d1 >> 16) & 0xffffu)     // tensor_dim1[31:16]
          |       ((tile_d0 & 0xffffu) << 16));     // tile_dim0 @ bits 127:112
    g1[4] = (int)(tile_d1 & 0xffffu);               // tile_dim1 (tile_dim2 = 0)
    g1[5] = (int)stride_d0;                         // tensor_dim0_stride[31:0]
    g1[6] = 0;                                      // stride[47:32] | dim1_stride lo
    g1[7] = 0;
    v4i z4 = (v4i)(0);
    v8i z8 = (v8i)(0);
    __builtin_amdgcn_tensor_load_to_lds(g0, g1, z4, z4, z8, /*cpol=*/0);
}

// ---------------------------------------------------------------------------
// K1: node degrees Dv, inv-sqrt, and XsT[c,n] = bf16( x[n,c] * Dv^{-1/2}[n] )
// one wave per node row n. grid (N/8, B), block 256.
// ---------------------------------------------------------------------------
__global__ void k1_dv_xst(const float* __restrict__ x, const float* __restrict__ H,
                          float* __restrict__ isDv, __bf16* __restrict__ XsT) {
    int wave = threadIdx.x >> 5, lane = threadIdx.x & 31;
    int n = blockIdx.x * 8 + wave;
    int b = blockIdx.y;
    const float* hrow = H + ((long)b * NN + n) * EE;
    float s = 0.f;
    for (int e = lane * 4; e < EE; e += 32 * 4) {
        float4 v = *(const float4*)(hrow + e);
        s += v.x + v.y + v.z + v.w;
    }
    #pragma unroll
    for (int off = 16; off; off >>= 1) s += __shfl_down(s, off, 32);
    float is = rsqrtf(__shfl(s, 0, 32) + EPSV);
    if (lane == 0) isDv[b * NN + n] = is;
    const float* xrow = x + ((long)b * NN + n) * CC;
    #pragma unroll
    for (int c = lane; c < CC; c += 32)
        XsT[((long)b * CC + c) * NN + n] = (__bf16)(xrow[c] * is);
}

// ---------------------------------------------------------------------------
// K2: hyperedge degrees De (column sums of H). grid (E/256, 16, B), block 256.
// De zeroed by hipMemsetAsync. eps folded in at use site.
// ---------------------------------------------------------------------------
__global__ void k2_de(const float* __restrict__ H, float* __restrict__ De) {
    int e = blockIdx.x * 256 + threadIdx.x;
    int b = blockIdx.z;
    const int rows = NN / 16;
    int n0 = blockIdx.y * rows;
    const float* p = H + ((long)b * NN + n0) * EE + e;
    float s = 0.f;
    for (int i = 0; i < rows; ++i) s += p[(long)i * EE];
    atomicAdd(&De[b * EE + e], s);
}

// ---------------------------------------------------------------------------
// K3: Ht[e,n] = bf16(H[n,e])  (LDS-tiled 32x32 transpose).
// grid (E/32, N/32, B), block (32,8).
// ---------------------------------------------------------------------------
__global__ void k3_ht(const float* __restrict__ H, __bf16* __restrict__ Ht) {
    __shared__ float t[32][33];
    int e0 = blockIdx.x * 32, n0 = blockIdx.y * 32, b = blockIdx.z;
    #pragma unroll
    for (int i = 0; i < 4; ++i) {
        int n = n0 + threadIdx.y + i * 8;
        t[threadIdx.y + i * 8][threadIdx.x] =
            H[((long)b * NN + n) * EE + e0 + threadIdx.x];
    }
    __syncthreads();
    #pragma unroll
    for (int i = 0; i < 4; ++i) {
        int e = e0 + threadIdx.y + i * 8;
        Ht[((long)b * EE + e) * NN + n0 + threadIdx.x] =
            (__bf16)t[threadIdx.x][threadIdx.y + i * 8];
    }
}

// ---------------------------------------------------------------------------
// K3b: W -> bf16. 16384 elements.
// ---------------------------------------------------------------------------
__global__ void k3b_w(const float* __restrict__ W, __bf16* __restrict__ Wb) {
    int i = blockIdx.x * 256 + threadIdx.x;
    if (i < CC * CC) Wb[i] = (__bf16)W[i];
}

// ---------------------------------------------------------------------------
// K4: GEMM1  T[e,c] = sum_n Ht[e,n] * Xs[n,c]   (K = N = 4096)
//     store  T2T[c,e] = bf16( T[e,c] / (De[e]+eps) )
// one wave per 16-row e-strip x full C=128. grid 128 blocks of 8 waves.
// B-panel (XsT, shared by all 8 waves) staged via TDM, double buffered.
// Dynamic LDS: 2 * CC*KCH*2 = 64 KB.
// ---------------------------------------------------------------------------
__global__ void k4_gemm1(const __bf16* __restrict__ Ht, const __bf16* __restrict__ XsT,
                         const float* __restrict__ De, __bf16* __restrict__ T2T) {
    extern __shared__ char smem[];
    const unsigned bufBytes = CC * KCH * 2;

    int w = threadIdx.x >> 5, lane = threadIdx.x & 31;
    int row = lane & 15, hi = lane >> 4;
    int strip = blockIdx.x * 8 + w;          // 0 .. B*E/16-1
    int b  = strip >> 7;                     // E/16 = 128 strips per batch
    int e0 = (strip & 127) << 4;

    const __bf16* Arow  = Ht + ((long)b * EE + e0 + row) * NN + hi * 8;
    const __bf16* Bbase = XsT + (long)b * CC * NN;   // [CC][NN] row-major
    bool w0 = (threadIdx.x < 32);

    v8f acc[8];
    #pragma unroll
    for (int t = 0; t < 8; ++t) acc[t] = (v8f)(0.f);

    const int NCHUNK = NN / KCH;
    if (w0)
        tdm_load_2d_bf16((unsigned)(unsigned long long)&smem[0], Bbase,
                         NN, CC, KCH, CC, NN);

    for (int ci = 0; ci < NCHUNK; ++ci) {
        if (w0) {
            if (ci + 1 < NCHUNK) {
                tdm_load_2d_bf16((unsigned)(unsigned long long)&smem[((ci + 1) & 1) * bufBytes],
                                 Bbase + (long)(ci + 1) * KCH,
                                 NN, CC, KCH, CC, NN);
                __builtin_amdgcn_s_wait_tensorcnt(1);   // chunk ci complete
            } else {
                __builtin_amdgcn_s_wait_tensorcnt(0);
            }
        }
        __syncthreads();
        const __bf16* bb = (const __bf16*)(smem + (ci & 1) * bufBytes); // [CC][KCH]
        const __bf16* ap = Arow + ci * KCH;
        #pragma unroll
        for (int kk = 0; kk < KCH; kk += 32) {
            v16bf a = load_a_frag(ap + kk);
            #pragma unroll
            for (int t = 0; t < 8; ++t) {
                v16bf bf = load_b_frag(bb + (t * 16 + row) * KCH + kk + hi * 8);
                acc[t] = wmma_bf16(a, bf, acc[t]);
            }
        }
        __syncthreads();   // protect buffer before next TDM overwrite
    }

    float rde[8];
    #pragma unroll
    for (int r = 0; r < 8; ++r)
        rde[r] = 1.0f / (De[b * EE + e0 + r + hi * 8] + EPSV);

    #pragma unroll
    for (int t = 0; t < 8; ++t) {
        int c = t * 16 + row;
        __bf16* outc = T2T + ((long)b * CC + c) * EE + e0;
        #pragma unroll
        for (int r = 0; r < 8; ++r)
            outc[r + hi * 8] = (__bf16)(acc[t][r] * rde[r]);
    }
}

// ---------------------------------------------------------------------------
// K5: fused GEMM2 + Dv^{-1/2} scale + GEMM3(+bias).
//   Y[n,c]  = sum_e H[n,e] * T2[e,c]          (K = E = 2048, A from f32 H)
//   Y      *= Dv^{-1/2}[n]
//   Out[n,o]= sum_c Y[n,c] * W[o,c] + bias[o] (K = C = 128, A re-staged via LDS)
// one wave per 16-row n-strip. grid (N/128, B), block 256.
// B-panel (T2T) staged via TDM. Dynamic LDS: 64 KB panels + 34 KB Y tiles.
// ---------------------------------------------------------------------------
__global__ void k5_fused(const float* __restrict__ H, const __bf16* __restrict__ T2T,
                         const float* __restrict__ isDv, const __bf16* __restrict__ Wb,
                         const float* __restrict__ bias, float* __restrict__ out) {
    extern __shared__ char smem[];
    const unsigned bufBytes = CC * KCH * 2;
    __bf16* yt = (__bf16*)(smem + 2 * bufBytes);     // [8][16][136] bf16

    int w = threadIdx.x >> 5, lane = threadIdx.x & 31;
    int row = lane & 15, hi = lane >> 4;
    int b  = blockIdx.y;
    int n0 = blockIdx.x * 128 + w * 16;

    const float*  Arow  = H + ((long)b * NN + n0 + row) * EE + hi * 8;
    const __bf16* Bbase = T2T + (long)b * CC * EE;   // [CC][EE] row-major
    bool w0 = (threadIdx.x < 32);

    v8f acc[8];
    #pragma unroll
    for (int t = 0; t < 8; ++t) acc[t] = (v8f)(0.f);

    const int NCHUNK = EE / KCH;
    if (w0)
        tdm_load_2d_bf16((unsigned)(unsigned long long)&smem[0], Bbase,
                         EE, CC, KCH, CC, EE);

    for (int ci = 0; ci < NCHUNK; ++ci) {
        if (w0) {
            if (ci + 1 < NCHUNK) {
                tdm_load_2d_bf16((unsigned)(unsigned long long)&smem[((ci + 1) & 1) * bufBytes],
                                 Bbase + (long)(ci + 1) * KCH,
                                 EE, CC, KCH, CC, EE);
                __builtin_amdgcn_s_wait_tensorcnt(1);
            } else {
                __builtin_amdgcn_s_wait_tensorcnt(0);
            }
        }
        __syncthreads();
        const __bf16* bb = (const __bf16*)(smem + (ci & 1) * bufBytes); // [CC][KCH]
        const float*  ap = Arow + ci * KCH;
        #pragma unroll
        for (int kk = 0; kk < KCH; kk += 32) {
            // A fragment straight from f32 H, converted to bf16
            union { v16bf v; __bf16 e[16]; } ua;
            const float* p = ap + kk;
            #pragma unroll
            for (int i = 0; i < 8; ++i) ua.e[i]     = (__bf16)p[i];
            #pragma unroll
            for (int i = 0; i < 8; ++i) ua.e[8 + i] = (__bf16)p[16 + i];
            #pragma unroll
            for (int t = 0; t < 8; ++t) {
                v16bf bf = load_b_frag(bb + (t * 16 + row) * KCH + kk + hi * 8);
                acc[t] = wmma_bf16(ua.v, bf, acc[t]);
            }
        }
        __syncthreads();
    }

    // scale rows by Dv^{-1/2}
    float isr[8];
    #pragma unroll
    for (int r = 0; r < 8; ++r)
        isr[r] = isDv[b * NN + n0 + r + hi * 8];
    #pragma unroll
    for (int t = 0; t < 8; ++t)
        #pragma unroll
        for (int r = 0; r < 8; ++r)
            acc[t][r] *= isr[r];

    // stage Y tile to LDS in bf16 (D-layout -> A-layout re-read), stride 136
    #pragma unroll
    for (int t = 0; t < 8; ++t)
        #pragma unroll
        for (int r = 0; r < 8; ++r)
            yt[((w * 16) + r + hi * 8) * 136 + t * 16 + row] = (__bf16)acc[t][r];
    __syncthreads();

    v8f acc2[8];
    #pragma unroll
    for (int t = 0; t < 8; ++t) acc2[t] = (v8f)(0.f);

    #pragma unroll
    for (int k0 = 0; k0 < CC; k0 += 32) {
        v16bf a = load_a_frag(&yt[(w * 16 + row) * 136 + k0 + hi * 8]);
        #pragma unroll
        for (int t = 0; t < 8; ++t) {
            v16bf bf = load_b_frag(Wb + ((long)(t * 16 + row)) * CC + k0 + hi * 8);
            acc2[t] = wmma_bf16(a, bf, acc2[t]);
        }
    }

    #pragma unroll
    for (int t = 0; t < 8; ++t) {
        float bs = bias[t * 16 + row];
        #pragma unroll
        for (int r = 0; r < 8; ++r) {
            int n = n0 + r + hi * 8;
            out[((long)b * NN + n) * CC + t * 16 + row] = acc2[t][r] + bs;
        }
    }
}

// ---------------------------------------------------------------------------
extern "C" void kernel_launch(void* const* d_in, const int* in_sizes, int n_in,
                              void* d_out, int out_size, void* d_ws, size_t ws_size,
                              hipStream_t stream) {
    const float* x = (const float*)d_in[0];
    const float* H = (const float*)d_in[1];
    const float* W = (const float*)d_in[2];
    const float* bvec = (const float*)d_in[3];
    float* out = (float*)d_out;

    // workspace carve-out (all sizes 256B-aligned)
    char* ws = (char*)d_ws;
    __bf16* Ht   = (__bf16*)ws;                 ws += (size_t)BB * EE * NN * 2;  // 128 MB
    __bf16* XsT  = (__bf16*)ws;                 ws += (size_t)BB * CC * NN * 2;  // 8 MB
    __bf16* T2T  = (__bf16*)ws;                 ws += (size_t)BB * CC * EE * 2;  // 4 MB
    float*  De   = (float*)ws;                  ws += (size_t)BB * EE * 4;       // 64 KB
    float*  isDv = (float*)ws;                  ws += (size_t)BB * NN * 4;       // 128 KB
    __bf16* Wb   = (__bf16*)ws;                 ws += (size_t)CC * CC * 2;       // 32 KB
    (void)ws_size; (void)in_sizes; (void)n_in; (void)out_size;

    const unsigned ldsK4 = 2u * CC * KCH * 2u;                 // 64 KB
    const unsigned ldsK5 = 2u * CC * KCH * 2u + 8u * 16u * 136u * 2u; // ~98 KB

    k1_dv_xst<<<dim3(NN / 8, BB), 256, 0, stream>>>(x, H, isDv, XsT);
    (void)hipMemsetAsync(De, 0, (size_t)BB * EE * 4, stream);
    k2_de<<<dim3(EE / 256, 16, BB), 256, 0, stream>>>(H, De);
    k3_ht<<<dim3(EE / 32, NN / 32, BB), dim3(32, 8), 0, stream>>>(H, Ht);
    k3b_w<<<dim3((CC * CC + 255) / 256), 256, 0, stream>>>(W, Wb);
    k4_gemm1<<<dim3(BB * EE / 16 / 8), 256, ldsK4, stream>>>(Ht, XsT, De, T2T);
    k5_fused<<<dim3(NN / 128, BB), 256, ldsK5, stream>>>(H, T2T, isDv, Wb, bvec, out);
}
